// AutoregressiveMlpMixer_61280593379794
// MI455X (gfx1250) — matrix operand order repaired and checked by reading, not compiled, not run on hardware
//
#include <hip/hip_runtime.h>
#include <hip/hip_bf16.h>
#include <cstdint>
#include <cstddef>

// ---------------- problem constants (from reference) ----------------
constexpr int Bc   = 64;
constexpr int CINc = 2;
constexpr int Hc   = 32;
constexpr int Wc   = 32;
constexpr int Nc   = 256;   // tokens
constexpr int Cc   = 768;   // hidden dim
constexpr int TOKc = 512;   // tokens mlp dim
constexpr int CHc  = 3072;  // channels mlp dim
constexpr int Lc   = 8;     // blocks
constexpr int Kcls = 2048;  // classes

// ---------------- WMMA vector types ----------------
typedef _Float16 v16h __attribute__((ext_vector_type(16)));
typedef _Float16 v8h  __attribute__((ext_vector_type(8)));
typedef float    v8f  __attribute__((ext_vector_type(8)));

union HF16x16 { v16h v; v8h h[2]; };

__device__ __forceinline__ float gelu_exact(float x) {
    return 0.5f * x * (1.0f + erff(x * 0.70710678118654752440f));
}

// =====================================================================
// Tiled WMMA GEMM:  Out[M,N] = act(A[M,K] @ Wt[N,K]^T + bias[N])
// A: f16 row-major (M,K).  Wt: f16 row-major (N,K) (pre-transposed).
// v_wmma_f32_16x16x32_f16, f32 accumulation.
// M%64==0, N%128==0, K%64==0.
// Block = 256 threads = 8 waves (4M x 2N); wave owns 16x64 strip.
// LDS double-buffered; staging via GLOBAL_LOAD_ASYNC_TO_LDS_B128
// (ASYNCcnt), one s_wait_asynccnt + barrier per K-stage.
// =====================================================================
constexpr int BM = 64, BN = 128, BK = 64;
constexpr int LDH = BK + 8;            // f16 row stride (144B, 16B-aligned)

template <typename OutT>
__global__ __launch_bounds__(256) void gemm_wmma(
    const _Float16* __restrict__ A, const _Float16* __restrict__ Wt,
    const float* __restrict__ bias, OutT* __restrict__ Out,
    int M, int N, int K, int gelu_flag)
{
    __shared__ __align__(16) _Float16 As[2][BM][LDH];
    __shared__ __align__(16) _Float16 Bs[2][BN][LDH];

    const int tid  = threadIdx.x;
    const int bm   = blockIdx.x * BM;
    const int bn   = blockIdx.y * BN;
    const int lane = tid & 31;
    const int wv   = tid >> 5;
    const int wm   = wv & 3;           // 0..3  M strip (16 rows)
    const int wn   = wv >> 2;          // 0..1  N strip (64 cols)
    const int lrow = lane & 15;
    const int g    = lane >> 4;        // lane group

    // staging coordinates: A chunk = 16 f16 (2 x b128), B chunk = 32 f16 (4 x b128)
    const int ra = tid >> 2;           // 0..63   A row
    const int ka = (tid & 3) << 4;     // 0,16,32,48
    const int rb = tid >> 1;           // 0..127  B row (output col)
    const int kb = (tid & 1) << 5;     // 0,32

    v8f acc[4] = {};

    // async global -> LDS staging (no VGPR data path, tracked by ASYNCcnt).
    // NB: the offset: immediate is added to BOTH the LDS and global address
    // (ISA 15.18.3 GLOBAL_LOAD_ASYNC_TO_LDS pseudocode).
    auto stage = [&](int k0, int pb) {
        unsigned lA = (unsigned)(uintptr_t)&As[pb][ra][ka];
        unsigned long long gA =
            (unsigned long long)(uintptr_t)(A + (size_t)(bm + ra) * K + k0 + ka);
        asm volatile(
            "global_load_async_to_lds_b128 %0, %1, off\n\t"
            "global_load_async_to_lds_b128 %0, %1, off offset:16"
            :: "v"(lA), "v"(gA) : "memory");
        unsigned lB = (unsigned)(uintptr_t)&Bs[pb][rb][kb];
        unsigned long long gB =
            (unsigned long long)(uintptr_t)(Wt + (size_t)(bn + rb) * K + k0 + kb);
        asm volatile(
            "global_load_async_to_lds_b128 %0, %1, off\n\t"
            "global_load_async_to_lds_b128 %0, %1, off offset:16\n\t"
            "global_load_async_to_lds_b128 %0, %1, off offset:32\n\t"
            "global_load_async_to_lds_b128 %0, %1, off offset:48"
            :: "v"(lB), "v"(gB) : "memory");
    };

    stage(0, 0);
    const int nstages = K / BK;
    for (int s = 0; s < nstages; ++s) {
        const int pb = s & 1;
        asm volatile("s_wait_asynccnt 0x0" ::: "memory");   // my tile landed in LDS
        __syncthreads();                                    // everyone's tile landed
        if (s + 1 < nstages) stage((s + 1) * BK, pb ^ 1);

        // ---- load ALL fragments into distinct registers (batched DS) ----
        HF16x16 a[2];
        {
            const _Float16* ap = &As[pb][wm * 16 + lrow][0];
            #pragma unroll
            for (int s2 = 0; s2 < 2; ++s2) {
                a[s2].h[0] = *(const v8h*)(ap + s2 * 32 + 8 * g);
                a[s2].h[1] = *(const v8h*)(ap + s2 * 32 + 16 + 8 * g);
            }
        }
        HF16x16 b[2][4];
        #pragma unroll
        for (int t = 0; t < 4; ++t) {
            const _Float16* bp = &Bs[pb][wn * 64 + t * 16 + lrow][0];
            #pragma unroll
            for (int s2 = 0; s2 < 2; ++s2) {
                b[s2][t].h[0] = *(const v8h*)(bp + s2 * 32 + 8 * g);
                b[s2][t].h[1] = *(const v8h*)(bp + s2 * 32 + 16 + 8 * g);
            }
        }
        // ---- 8 WMMAs per stage per wave ----
        #pragma unroll
        for (int s2 = 0; s2 < 2; ++s2)
            #pragma unroll
            for (int t = 0; t < 4; ++t)
                acc[t] = __builtin_amdgcn_wmma_f32_16x16x32_f16(
                    false, a[s2].v, false, b[s2][t].v, (short)0, acc[t], false, false);

        // ---- scheduling pipeline hint: issue all 20 DS reads (mask 0x100)
        //      before the 8 WMMAs (MFMA/WMMA mask 0x8) so dscnt waits are
        //      partial instead of a full drain per WMMA ----
        __builtin_amdgcn_sched_group_barrier(0x100, 20, 0);
        __builtin_amdgcn_sched_group_barrier(0x008, 8, 0);
    }

    // epilogue: D layout = VGPR r: lanes0-15 M=r, lanes16-31 M=8+r; N=lane%16
    const int rowBase = bm + wm * 16 + 8 * g;
    #pragma unroll
    for (int t = 0; t < 4; ++t) {
        const int col = bn + wn * 64 + t * 16 + lrow;
        const float bv = bias ? bias[col] : 0.0f;
        #pragma unroll
        for (int r = 0; r < 8; ++r) {
            float v = acc[t][r] + bv;
            if (gelu_flag) v = gelu_exact(v);
            Out[(size_t)(rowBase + r) * N + col] = (OutT)v;
        }
    }
}

// =====================================================================
// Weight prep: w (K,N) f32  ->  wt (N,K) f16   (32x32 LDS transpose)
// grid = (N/32, K/32), block = 256
// =====================================================================
__global__ __launch_bounds__(256) void wprep_kernel(
    const float* __restrict__ w, _Float16* __restrict__ wt, int K, int N)
{
    __shared__ float tile[32][33];
    const int n0 = blockIdx.x * 32;
    const int k0 = blockIdx.y * 32;
    const int tx = threadIdx.x & 31;
    const int ty = threadIdx.x >> 5;
    #pragma unroll
    for (int i = 0; i < 32; i += 8)
        tile[ty + i][tx] = w[(size_t)(k0 + ty + i) * N + (n0 + tx)];
    __syncthreads();
    #pragma unroll
    for (int i = 0; i < 32; i += 8)
        wt[(size_t)(n0 + ty + i) * K + (k0 + tx)] = (_Float16)tile[tx][ty + i];
}

// =====================================================================
// Patch-embed stem: (B,CIN,32,32) --2x2/s2 conv--> x (B,N,C) f16
// =====================================================================
__global__ __launch_bounds__(256) void stem_kernel(
    const float* __restrict__ in, const float* __restrict__ w,
    const float* __restrict__ bias, _Float16* __restrict__ out)
{
    int idx = blockIdx.x * blockDim.x + threadIdx.x;
    if (idx >= Bc * Nc * Cc) return;
    int c = idx % Cc;
    int n = (idx / Cc) % Nc;
    int b = idx / (Cc * Nc);
    int hp = n >> 4, wp = n & 15;
    float acc = bias[c];
    #pragma unroll
    for (int ci = 0; ci < CINc; ++ci)
        #pragma unroll
        for (int ph = 0; ph < 2; ++ph)
            #pragma unroll
            for (int pw = 0; pw < 2; ++pw)
                acc += w[((c * CINc + ci) * 2 + ph) * 2 + pw] *
                       in[((size_t)(b * CINc + ci) * Hc + (hp * 2 + ph)) * Wc + (wp * 2 + pw)];
    out[idx] = (_Float16)acc;
}

// =====================================================================
// LayerNorm over width=768 (=3*256), f16 in/out, f32 math.
// transpose_out==0: out row-major (rows,768)
// transpose_out==1: rows are (b,n); out[b*768*tokN + c*tokN + n]
// =====================================================================
__global__ __launch_bounds__(256) void ln_kernel(
    const _Float16* __restrict__ in, _Float16* __restrict__ out,
    const float* __restrict__ gamma, const float* __restrict__ beta,
    int width, int transpose_out, int tokN)
{
    __shared__ float red[256];
    const int row = blockIdx.x;
    const int tid = threadIdx.x;
    const _Float16* x = in + (size_t)row * width;

    float v[3];
    float s = 0.f;
    #pragma unroll
    for (int i = 0; i < 3; ++i) { v[i] = (float)x[tid + i * 256]; s += v[i]; }
    red[tid] = s; __syncthreads();
    for (int off = 128; off > 0; off >>= 1) {
        if (tid < off) red[tid] += red[tid + off];
        __syncthreads();
    }
    const float mean = red[0] / (float)width;
    __syncthreads();

    float s2 = 0.f;
    #pragma unroll
    for (int i = 0; i < 3; ++i) { float d = v[i] - mean; s2 += d * d; }
    red[tid] = s2; __syncthreads();
    for (int off = 128; off > 0; off >>= 1) {
        if (tid < off) red[tid] += red[tid + off];
        __syncthreads();
    }
    const float rstd = rsqrtf(red[0] / (float)width + 1e-5f);

    if (!transpose_out) {
        _Float16* o = out + (size_t)row * width;
        #pragma unroll
        for (int i = 0; i < 3; ++i) {
            int c = tid + i * 256;
            o[c] = (_Float16)((v[i] - mean) * rstd * gamma[c] + beta[c]);
        }
    } else {
        int b = row / tokN, n = row % tokN;
        _Float16* o = out + (size_t)b * width * tokN + n;
        #pragma unroll
        for (int i = 0; i < 3; ++i) {
            int c = tid + i * 256;
            o[(size_t)c * tokN] = (_Float16)((v[i] - mean) * rstd * gamma[c] + beta[c]);
        }
    }
}

// =====================================================================
// In-place reverse (suffix) cumsum over 256 tokens, f16 storage/f32 math
// =====================================================================
__global__ __launch_bounds__(256) void revcumsum_kernel(_Float16* __restrict__ data)
{
    __shared__ float s[256];
    const int tid = threadIdx.x;
    _Float16* p = data + (size_t)blockIdx.x * 256;
    s[tid] = (float)p[255 - tid];
    __syncthreads();
    #pragma unroll
    for (int off = 1; off < 256; off <<= 1) {
        float t = (tid >= off) ? s[tid - off] : 0.f;
        __syncthreads();
        s[tid] += t;
        __syncthreads();
    }
    p[255 - tid] = (_Float16)s[tid];
}

// =====================================================================
// Transpose (B,C,N) -> (B,N,C), f16, 32x32 LDS tiles
// grid = (N/32, C/32, B)
// =====================================================================
__global__ __launch_bounds__(256) void transpose_cn_nc(
    const _Float16* __restrict__ in, _Float16* __restrict__ out, int Cdim, int Ndim)
{
    __shared__ _Float16 tile[32][33];
    const int b  = blockIdx.z;
    const int n0 = blockIdx.x * 32;
    const int c0 = blockIdx.y * 32;
    const int tx = threadIdx.x & 31;
    const int ty = threadIdx.x >> 5;
    const _Float16* src = in + (size_t)b * Cdim * Ndim;
    #pragma unroll
    for (int i = 0; i < 32; i += 8)
        tile[ty + i][tx] = src[(size_t)(c0 + ty + i) * Ndim + (n0 + tx)];
    __syncthreads();
    _Float16* dst = out + (size_t)b * Ndim * Cdim;
    #pragma unroll
    for (int i = 0; i < 32; i += 8)
        dst[(size_t)(n0 + ty + i) * Cdim + (c0 + tx)] = tile[tx][ty + i];
}

// =====================================================================
// Mean over tokens: in (B,N,C) f16 -> out (B,C) f16
// =====================================================================
__global__ __launch_bounds__(256) void mean_tokens_kernel(
    const _Float16* __restrict__ in, _Float16* __restrict__ out)
{
    int idx = blockIdx.x * blockDim.x + threadIdx.x;
    if (idx >= Bc * Cc) return;
    int b = idx / Cc, c = idx % Cc;
    const _Float16* p = in + (size_t)b * Nc * Cc + c;
    float s = 0.f;
    for (int n = 0; n < Nc; ++n) s += (float)p[(size_t)n * Cc];
    out[idx] = (_Float16)(s * (1.0f / (float)Nc));
}

// =====================================================================
// Host-side orchestration
// =====================================================================
extern "C" void kernel_launch(void* const* d_in, const int* in_sizes, int n_in,
                              void* d_out, int out_size, void* d_ws, size_t ws_size,
                              hipStream_t stream)
{
    (void)in_sizes; (void)n_in; (void)out_size; (void)ws_size;

    const float* inputs = (const float*)d_in[0];
    const float* stem_w = (const float*)d_in[1];
    const float* stem_b = (const float*)d_in[2];
    const float* ln1_g  = (const float*)d_in[3];
    const float* ln1_b  = (const float*)d_in[4];
    const float* tok_w1 = (const float*)d_in[5];
    const float* tok_b1 = (const float*)d_in[6];
    const float* tok_w2 = (const float*)d_in[7];
    const float* tok_b2 = (const float*)d_in[8];
    const float* ln2_g  = (const float*)d_in[9];
    const float* ln2_b  = (const float*)d_in[10];
    const float* ch_w1  = (const float*)d_in[11];
    const float* ch_b1  = (const float*)d_in[12];
    const float* ch_w2  = (const float*)d_in[13];
    const float* ch_b2  = (const float*)d_in[14];
    const float* lnf_g  = (const float*)d_in[15];
    const float* lnf_b  = (const float*)d_in[16];
    const float* head_w = (const float*)d_in[17];
    const float* head_b = (const float*)d_in[18];
    float* out = (float*)d_out;

    // ---- workspace layout (f16 elements; every offset multiple of 8) ----
    const size_t SZ_BNC = (size_t)Bc * Nc * Cc;        // 12,582,912
    const size_t SZ_BIG = (size_t)Bc * Nc * CHc;       // 50,331,648
    _Float16* ws      = (_Float16*)d_ws;
    _Float16* bufX    = ws;                            // x        (B,N,C)
    _Float16* bufT    = bufX + SZ_BNC;                 // yT       (B,C,N)
    _Float16* bufD    = bufT + SZ_BNC;                 // y        (B,N,C)
    _Float16* bufBig  = bufD + SZ_BNC;                 // (B,C,TOK)/(B,N,CH)
    _Float16* bufMean = bufBig + SZ_BIG;               // (B,C)
    _Float16* wtTok1  = bufMean + (size_t)Bc * Cc;     // 8 x (512,256)
    _Float16* wtTok2  = wtTok1 + (size_t)Lc * TOKc * Nc;   // 8 x (256,512)
    _Float16* wtCh1   = wtTok2 + (size_t)Lc * Nc * TOKc;   // 8 x (3072,768)
    _Float16* wtCh2   = wtCh1 + (size_t)Lc * CHc * Cc;     // 8 x (768,3072)
    _Float16* wtHead  = wtCh2 + (size_t)Lc * Cc * CHc;     // (2048,768)

    // ---- 0) weight convert+transpose to f16 [N][K]
    for (int l = 0; l < Lc; ++l) {
        wprep_kernel<<<dim3(TOKc / 32, Nc / 32), 256, 0, stream>>>(
            tok_w1 + (size_t)l * Nc * TOKc, wtTok1 + (size_t)l * TOKc * Nc, Nc, TOKc);
        wprep_kernel<<<dim3(Nc / 32, TOKc / 32), 256, 0, stream>>>(
            tok_w2 + (size_t)l * TOKc * Nc, wtTok2 + (size_t)l * Nc * TOKc, TOKc, Nc);
        wprep_kernel<<<dim3(CHc / 32, Cc / 32), 256, 0, stream>>>(
            ch_w1 + (size_t)l * Cc * CHc, wtCh1 + (size_t)l * CHc * Cc, Cc, CHc);
        wprep_kernel<<<dim3(Cc / 32, CHc / 32), 256, 0, stream>>>(
            ch_w2 + (size_t)l * CHc * Cc, wtCh2 + (size_t)l * Cc * CHc, CHc, Cc);
    }
    wprep_kernel<<<dim3(Kcls / 32, Cc / 32), 256, 0, stream>>>(head_w, wtHead, Cc, Kcls);

    // ---- 1) patch-embed stem -> x (B,N,C) f16
    {
        int total = Bc * Nc * Cc;
        stem_kernel<<<(total + 255) / 256, 256, 0, stream>>>(inputs, stem_w, stem_b, bufX);
    }

    // ---- 2) mixer blocks
    const int MtokGemm = Bc * Cc;   // 49152
    const int MchGemm  = Bc * Nc;   // 16384
    for (int l = 0; l < Lc; ++l) {
        ln_kernel<<<Bc * Nc, 256, 0, stream>>>(bufX, bufT, ln1_g + l * Cc, ln1_b + l * Cc,
                                               Cc, 1, Nc);
        revcumsum_kernel<<<Bc * Cc, 256, 0, stream>>>(bufT);
        gemm_wmma<_Float16><<<dim3(MtokGemm / BM, TOKc / BN), 256, 0, stream>>>(
            bufT, wtTok1 + (size_t)l * TOKc * Nc, tok_b1 + l * TOKc, bufBig,
            MtokGemm, TOKc, Nc, 1);
        gemm_wmma<_Float16><<<dim3(MtokGemm / BM, Nc / BN), 256, 0, stream>>>(
            bufBig, wtTok2 + (size_t)l * Nc * TOKc, tok_b2 + l * Nc, bufT,
            MtokGemm, Nc, TOKc, 0);
        transpose_cn_nc<<<dim3(Nc / 32, Cc / 32, Bc), 256, 0, stream>>>(bufT, bufD, Cc, Nc);
        ln_kernel<<<Bc * Nc, 256, 0, stream>>>(bufD, bufD, ln2_g + l * Cc, ln2_b + l * Cc,
                                               Cc, 0, Nc);
        gemm_wmma<_Float16><<<dim3(MchGemm / BM, CHc / BN), 256, 0, stream>>>(
            bufD, wtCh1 + (size_t)l * CHc * Cc, ch_b1 + l * CHc, bufBig,
            MchGemm, CHc, Cc, 1);
        gemm_wmma<_Float16><<<dim3(MchGemm / BM, Cc / BN), 256, 0, stream>>>(
            bufBig, wtCh2 + (size_t)l * Cc * CHc, ch_b2 + l * Cc, bufX,
            MchGemm, Cc, CHc, 0);
    }

    // ---- 3) final LN (in place on x)
    ln_kernel<<<Bc * Nc, 256, 0, stream>>>(bufX, bufX, lnf_g, lnf_b, Cc, 0, Nc);

    // ---- 4) mean over tokens -> (B,C)
    mean_tokens_kernel<<<(Bc * Cc + 255) / 256, 256, 0, stream>>>(bufX, bufMean);

    // ---- 5) classifier head: (64,768) @ (768,2048) + b -> d_out (f32)
    gemm_wmma<float><<<dim3(Bc / BM, Kcls / BN), 256, 0, stream>>>(
        bufMean, wtHead, head_b, out, Bc, Kcls, Cc, 0);
}
